// ACTP_30167850287458
// MI455X (gfx1250) — matrix operand order repaired and checked by reading, not compiled, not run on hardware
//
#include <hip/hip_runtime.h>

typedef __attribute__((ext_vector_type(16))) __bf16 bf16x16;
typedef __attribute__((ext_vector_type(8)))  __bf16 bf16x8;
typedef __attribute__((ext_vector_type(8)))  float  f32x8;

#define BATCH   8192
#define BM      32
#define NBLK    (BATCH / BM)     // 256 workgroups
#define TSTEPS  29               // T-1
#define CTX     10

// ---------------- weight packing (B-operand tiles for v_wmma_f32_16x16x32_bf16) --
// Packed layout: element index = ((nt*KC + kc)*32 + lane)*16 + j
// lane l supplies 16 contiguous-K bf16 for column n = nt*16 + (l&15),
// starting at k = kc*32 + (l>>4)*16   (contiguous k order, same as load_A_frag)
// Gate-padded N: packed row ntp -> gate g = ntp/NTPG, local row nr; orig n = g*NPG+nr.
__global__ void pack_w_kernel(const float* __restrict__ W, unsigned short* __restrict__ outp,
                              int NPG, int K, int NTPG, int KC, int total) {
  int idx = blockIdx.x * blockDim.x + threadIdx.x;
  if (idx >= total) return;
  int j    = idx & 15;
  int l    = (idx >> 4) & 31;
  int tile = idx >> 9;
  int kc   = tile % KC;
  int ntp  = tile / KC;
  int g    = ntp / NTPG;
  int jt   = ntp % NTPG;
  int nr   = jt * 16 + (l & 15);
  int k    = kc * 32 + ((l >> 4) << 4) + j;
  float v = 0.0f;
  if (nr < NPG && k < K) v = W[(g * NPG + nr) * K + k];
  __bf16 b = (__bf16)v;
  outp[idx] = *(unsigned short*)&b;
}

__global__ void prep_bias_kernel(const float* __restrict__ a, const float* __restrict__ b,
                                 float* __restrict__ outp, int NPG, int NPAD, int gates) {
  int i = blockIdx.x * blockDim.x + threadIdx.x;
  if (i >= gates * NPAD) return;
  int g = i / NPAD, nr = i % NPAD;
  float v = 0.0f;
  if (nr < NPG) { int n = g * NPG + nr; v = a[n]; if (b) v += b[n]; }
  outp[i] = v;
}

// ---------------- device helpers -------------------------------------------------
__device__ __forceinline__ float sigf(float x) { return 1.0f / (1.0f + __expf(-x)); }

__device__ __forceinline__ f32x8 wmma_bf16(bf16x16 a, bf16x16 b, f32x8 c) {
  return __builtin_amdgcn_wmma_f32_16x16x32_bf16(false, a, false, b, (short)0, c, false, false);
}

// A-fragment 16x32 bf16 from LDS activation buffer (row-major, leading dim ld).
// lane l: m = mt*16 + (l&15); 16 contiguous elements at k = (l>>4)*16.
__device__ __forceinline__ bf16x16 load_A_frag(const __bf16* act, int ld, int mt, int kbase, int lane) {
  int m = mt * 16 + (lane & 15);
  int r = (lane >> 4) << 4;
  const __bf16* p = act + m * ld + kbase;
  union { bf16x16 v; bf16x8 h[2]; } u;
  u.h[0] = *(const bf16x8*)(p + r);
  u.h[1] = *(const bf16x8*)(p + r + 8);
  return u.v;
}

// B-fragment 32x16 bf16 from packed global weights (L2-resident): 32B/lane contiguous.
__device__ __forceinline__ bf16x16 load_B_frag(const __bf16* pack, int nt, int kc, int KC, int lane) {
  return *(const bf16x16*)(pack + ((((nt * KC) + kc) * 32 + lane) << 4));
}

// One LSTM layer: gates = X@Wx^T + H@Wh^T + b; fused in-register nonlinearity.
// Packed weights hold 4 gate groups of 13 n-tiles each (N padded 200->208).
template <int KCX, int KCH>
__device__ void lstm_layer(const __bf16* X, int ldx, const __bf16* PX,
                           const __bf16* Hin, int ldh, const __bf16* PH,
                           const float* biasp, float* cbuf,
                           __bf16* hout, int ldo, __bf16* hout2, int ldo2,
                           int wave, int lane) {
  for (int p = wave; p < 26; p += 8) {
    int mt = p & 1, j = p >> 1;
    f32x8 ai = {0,0,0,0,0,0,0,0}, af = ai, ag = ai, ao = ai;
#pragma unroll
    for (int kc = 0; kc < KCX; ++kc) {
      bf16x16 a = load_A_frag(X, ldx, mt, kc * 32, lane);
      ai = wmma_bf16(a, load_B_frag(PX,      j, kc, KCX, lane), ai);
      af = wmma_bf16(a, load_B_frag(PX, 13 + j, kc, KCX, lane), af);
      ag = wmma_bf16(a, load_B_frag(PX, 26 + j, kc, KCX, lane), ag);
      ao = wmma_bf16(a, load_B_frag(PX, 39 + j, kc, KCX, lane), ao);
    }
#pragma unroll
    for (int kc = 0; kc < KCH; ++kc) {
      bf16x16 a = load_A_frag(Hin, ldh, mt, kc * 32, lane);
      ai = wmma_bf16(a, load_B_frag(PH,      j, kc, KCH, lane), ai);
      af = wmma_bf16(a, load_B_frag(PH, 13 + j, kc, KCH, lane), af);
      ag = wmma_bf16(a, load_B_frag(PH, 26 + j, kc, KCH, lane), ag);
      ao = wmma_bf16(a, load_B_frag(PH, 39 + j, kc, KCH, lane), ao);
    }
    int n = j * 16 + (lane & 15);                 // 0..207 (gate-padded)
    float bi = biasp[n], bf_ = biasp[208 + n], bg = biasp[416 + n], bo = biasp[624 + n];
    int mbase = mt * 16 + ((lane >> 4) << 3);
#pragma unroll
    for (int r = 0; r < 8; ++r) {
      float iv = sigf(ai[r] + bi);
      float fv = sigf(af[r] + bf_);
      float gv = tanhf(ag[r] + bg);
      float ov = sigf(ao[r] + bo);
      int ci = (mbase + r) * 208 + n;
      float cn = fv * cbuf[ci] + iv * gv;
      cbuf[ci] = cn;
      float hv = ov * tanhf(cn);
      hout[(mbase + r) * ldo + n] = (__bf16)hv;
      if (hout2 && n < 200) hout2[(mbase + r) * ldo2 + n] = (__bf16)hv;  // concat copy
    }
  }
}

template <int KC, int NT>
__device__ void fc_layer_tanh(const __bf16* X, int ldx, const __bf16* P, const float* biasp,
                              __bf16* outb, int ldo, int wave, int lane) {
  for (int p = wave; p < 2 * NT; p += 8) {
    int mt = p & 1, nt = p >> 1;
    f32x8 acc = {0,0,0,0,0,0,0,0};
#pragma unroll
    for (int kc = 0; kc < KC; ++kc) {
      bf16x16 a = load_A_frag(X, ldx, mt, kc * 32, lane);
      acc = wmma_bf16(a, load_B_frag(P, nt, kc, KC, lane), acc);
    }
    int n = nt * 16 + (lane & 15);
    float bv = biasp[n];
    int mbase = mt * 16 + ((lane >> 4) << 3);
#pragma unroll
    for (int r = 0; r < 8; ++r)
      outb[(mbase + r) * ldo + n] = (__bf16)tanhf(acc[r] + bv);
  }
}

// ---------------- main persistent kernel: 29 sequential steps per batch slice ----
__global__ __launch_bounds__(256) void actp_main(
    const float* __restrict__ tactiles, const float* __restrict__ actions,
    const unsigned short* wih1p_, const unsigned short* whh1p_,
    const unsigned short* wih2p_, const unsigned short* whh2p_,
    const unsigned short* fc1p_,  const unsigned short* fc2p_,
    const float* bias1p, const float* bias2p, const float* fc1bp, const float* fc2bp,
    float* __restrict__ out) {
  extern __shared__ char smem[];
  float*  c1  = (float*)smem;               // [32][208]
  float*  c2  = c1 + 32 * 208;              // [32][208]
  __bf16* h1A = (__bf16*)(c2 + 32 * 208);   // [32][224]
  __bf16* h1B = h1A + 32 * 224;
  __bf16* h2A = h1B + 32 * 224;
  __bf16* h2B = h2A + 32 * 224;
  __bf16* xb  = h2B + 32 * 224;             // [32][64]  x = [tactile|act_next|state]
  __bf16* xc  = xb + 32 * 64;               // [32][256] fc1 input = [h2|tactile|pad]
  __bf16* o3  = xc + 32 * 256;              // [32][224]

  const __bf16* wih1p = (const __bf16*)wih1p_;
  const __bf16* whh1p = (const __bf16*)whh1p_;
  const __bf16* wih2p = (const __bf16*)wih2p_;
  const __bf16* whh2p = (const __bf16*)whh2p_;
  const __bf16* fc1p  = (const __bf16*)fc1p_;
  const __bf16* fc2p  = (const __bf16*)fc2p_;

  int tid = threadIdx.x, wave = tid >> 5, lane = tid & 31;
  int bg0 = blockIdx.x * BM;

  for (int i = tid; i < 32 * 208; i += 256) { c1[i] = 0.f; c2[i] = 0.f; }
  for (int i = tid; i < 32 * 224; i += 256) {
    h1A[i] = (__bf16)0.f; h1B[i] = (__bf16)0.f;
    h2A[i] = (__bf16)0.f; h2B[i] = (__bf16)0.f;
    o3[i]  = (__bf16)0.f;
  }
  for (int i = tid; i < 32 * 256; i += 256) xc[i] = (__bf16)0.f;
  for (int i = tid; i < 32 * 64;  i += 256) xb[i] = (__bf16)0.f;
  __syncthreads();
  // state = actions[0], constant per step (x cols 54..59)
  for (int i = tid; i < 32 * 6; i += 256) {
    int b = i / 6, j = i % 6;
    xb[b * 64 + 54 + j] = (__bf16)actions[(bg0 + b) * 6 + j];
  }
  __syncthreads();

  for (int t = 0; t < TSTEPS; ++t) {
    // ---- stage inputs for this step ----
    if (t < CTX) {  // ground-truth tactile
      for (int i = tid; i < 32 * 48; i += 256) {
        int b = i / 48, j = i % 48;
        float v = tactiles[(t * BATCH + bg0 + b) * 48 + j];
        xb[b * 64 + j]        = (__bf16)v;
        xc[b * 256 + 200 + j] = (__bf16)v;
      }
    }
    for (int i = tid; i < 32 * 6; i += 256) {
      int b = i / 6, j = i % 6;
      xb[b * 64 + 48 + j] = (__bf16)actions[((t + 1) * BATCH + bg0 + b) * 6 + j];
    }
    __syncthreads();

    const __bf16* h1r = (t & 1) ? h1B : h1A;  __bf16* h1w = (t & 1) ? h1A : h1B;
    const __bf16* h2r = (t & 1) ? h2B : h2A;  __bf16* h2w = (t & 1) ? h2A : h2B;

    // ---- LSTM layer 1: x[32,64] + h1[32,224] ----
    lstm_layer<2, 7>(xb, 64, wih1p, h1r, 224, whh1p, bias1p, c1,
                     h1w, 224, (__bf16*)nullptr, 0, wave, lane);
    __syncthreads();

    // ---- LSTM layer 2: h1_new + h2_old; h2_new also copied into xc[:,0:200] ----
    lstm_layer<7, 7>(h1w, 224, wih2p, h2r, 224, whh2p, bias2p, c2,
                     h2w, 224, xc, 256, wave, lane);
    __syncthreads();

    // ---- fc1: tanh(concat(h2,new tactile)[32,256] @ fc1^T) -> o3 ----
    fc_layer_tanh<8, 13>(xc, 256, fc1p, fc1bp, o3, 224, wave, lane);
    __syncthreads();

    // ---- fc2: tanh(o3 @ fc2^T) -> out4; write output + feed back ----
    for (int p = wave; p < 6; p += 8) {
      int mt = p & 1, nt = p >> 1;
      f32x8 acc = {0,0,0,0,0,0,0,0};
#pragma unroll
      for (int kc = 0; kc < 7; ++kc) {
        bf16x16 a = load_A_frag(o3, 224, mt, kc * 32, lane);
        acc = wmma_bf16(a, load_B_frag(fc2p, nt, kc, 7, lane), acc);
      }
      int n = nt * 16 + (lane & 15);     // 0..47
      float bv = fc2bp[n];
      int mbase = mt * 16 + ((lane >> 4) << 3);
#pragma unroll
      for (int r = 0; r < 8; ++r) {
        float v = tanhf(acc[r] + bv);
        int b = mbase + r;
        xb[b * 64 + n]        = (__bf16)v;   // autoregressive feedback
        xc[b * 256 + 200 + n] = (__bf16)v;
        if (t >= CTX - 1)
          out[((t - (CTX - 1)) * BATCH + bg0 + b) * 48 + n] = v;
      }
    }
    __syncthreads();
  }
}

// ---------------- host launch ----------------------------------------------------
extern "C" void kernel_launch(void* const* d_in, const int* in_sizes, int n_in,
                              void* d_out, int out_size, void* d_ws, size_t ws_size,
                              hipStream_t stream) {
  const float* tactiles = (const float*)d_in[0];
  const float* actions  = (const float*)d_in[1];
  const float* W_ih1 = (const float*)d_in[2];
  const float* W_hh1 = (const float*)d_in[3];
  const float* b_ih1 = (const float*)d_in[4];
  const float* b_hh1 = (const float*)d_in[5];
  const float* W_ih2 = (const float*)d_in[6];
  const float* W_hh2 = (const float*)d_in[7];
  const float* b_ih2 = (const float*)d_in[8];
  const float* b_hh2 = (const float*)d_in[9];
  const float* fc1_w = (const float*)d_in[10];
  const float* fc1_b = (const float*)d_in[11];
  const float* fc2_w = (const float*)d_in[12];
  const float* fc2_b = (const float*)d_in[13];

  unsigned short* ws = (unsigned short*)d_ws;
  unsigned short* wih1p = ws;                       // 4*13*2*512 = 53248
  unsigned short* whh1p = wih1p + 53248;            // 4*13*7*512 = 186368
  unsigned short* wih2p = whh1p + 186368;
  unsigned short* whh2p = wih2p + 186368;
  unsigned short* fc1p  = whh2p + 186368;           // 1*13*8*512 = 53248
  unsigned short* fc2p  = fc1p  + 53248;            // 1*3*7*512  = 10752
  float* fbase  = (float*)(fc2p + 10752);
  float* bias1p = fbase;           // 4*208
  float* bias2p = fbase + 832;     // 4*208
  float* fc1bp  = fbase + 1664;    // 208
  float* fc2bp  = fbase + 1872;    // 48

  auto pack = [&](const float* W, unsigned short* o, int NPG, int K, int NTPG, int KC, int gates) {
    int total = gates * NTPG * KC * 512;
    pack_w_kernel<<<(total + 255) / 256, 256, 0, stream>>>(W, o, NPG, K, NTPG, KC, total);
  };
  pack(W_ih1, wih1p, 200,  60, 13, 2, 4);
  pack(W_hh1, whh1p, 200, 200, 13, 7, 4);
  pack(W_ih2, wih2p, 200, 200, 13, 7, 4);
  pack(W_hh2, whh2p, 200, 200, 13, 7, 4);
  pack(fc1_w, fc1p,  200, 248, 13, 8, 1);
  pack(fc2_w, fc2p,   48, 200,  3, 7, 1);

  prep_bias_kernel<<<4, 256, 0, stream>>>(b_ih1, b_hh1, bias1p, 200, 208, 4);
  prep_bias_kernel<<<4, 256, 0, stream>>>(b_ih2, b_hh2, bias2p, 200, 208, 4);
  prep_bias_kernel<<<1, 256, 0, stream>>>(fc1_b, (const float*)nullptr, fc1bp, 200, 208, 1);
  prep_bias_kernel<<<1, 256, 0, stream>>>(fc2_b, (const float*)nullptr, fc2bp, 48, 48, 1);

  // LDS: 2*32*208*4 + 4*32*224*2 + 32*64*2 + 32*256*2 + 32*224*2 = 145408 bytes
  size_t smem = 2 * 32 * 208 * 4 + 4 * 32 * 224 * 2 + 32 * 64 * 2 + 32 * 256 * 2 + 32 * 224 * 2;
  actp_main<<<NBLK, 256, smem, stream>>>(
      tactiles, actions, wih1p, whh1p, wih2p, whh2p, fc1p, fc2p,
      bias1p, bias2p, fc1bp, fc2bp, (float*)d_out);
}
